// AlignmentEncoder_27968827031686
// MI455X (gfx1250) — compile-verified, matrix operand order checked by reading, not run on hardware
//
#include <hip/hip_runtime.h>
#include <hip/hip_bf16.h>

// ---------------------------------------------------------------------------
// AlignmentEncoder for MI455X (gfx1250, wave32, WMMA + async global->LDS DMA).
// f16 WMMA inputs + f32 accumulation; activations t-major [B][T][C] so every
// fragment is two 128-bit LDS loads and every epilogue store is one b128.
// LDS staging uses GLOBAL_LOAD_ASYNC_TO_LDS_B128 (ASYNCcnt) when available.
// ---------------------------------------------------------------------------

typedef __attribute__((ext_vector_type(16))) _Float16 v16h;
typedef __attribute__((ext_vector_type(8)))  _Float16 v8h;
typedef __attribute__((ext_vector_type(8)))  float    v8f;
typedef int v4i __attribute__((vector_size(16)));   // matches builtin param type

#define TEMP 0.0005f

#define BATCH 32
#define C_MEL 80
#define C_TXT 512
#define C_ATT 80
#define T1    1600
#define T2    400

#if __has_builtin(__builtin_amdgcn_global_load_async_to_lds_b128)
#define ASYNC_LDS 1
#endif

// 16-byte global->LDS copy: async DMA on CDNA5, sync fallback otherwise.
__device__ __forceinline__ void cp_b128(void* lds, const void* g) {
#ifdef ASYNC_LDS
  __builtin_amdgcn_global_load_async_to_lds_b128(
      (__attribute__((address_space(1))) v4i*)g,
      (__attribute__((address_space(3))) v4i*)lds, 0, 0);
#else
  *(v8h*)lds = *(const v8h*)g;
#endif
}

__device__ __forceinline__ void zero_b128(void* lds) {
  v8h z = {};
  *(v8h*)lds = z;
}

__device__ __forceinline__ void wait_async() {
#ifdef ASYNC_LDS
#if __has_builtin(__builtin_amdgcn_s_wait_asynccnt)
  __builtin_amdgcn_s_wait_asynccnt(0);
#else
  asm volatile("s_wait_asynccnt 0" ::: "memory");
#endif
#endif
}

// ---------------------------------------------------------------------------
// Speaker projection: kbias[b][c] = spk[b]·Wks[c] + bks[c] (c<512),
//                     qbias[b][c] = spk[b]·Wqs[c] + bqs[c] (c<80)
// ---------------------------------------------------------------------------
__global__ void spk_proj_kernel(const float* __restrict__ spk,
                                const float* __restrict__ Wks, const float* __restrict__ bks,
                                const float* __restrict__ Wqs, const float* __restrict__ bqs,
                                float* __restrict__ kbias, float* __restrict__ qbias) {
  int b = blockIdx.y;
  int c = blockIdx.x * blockDim.x + threadIdx.x;
  const float* s = spk + (size_t)b * C_TXT;
  if (c < C_TXT) {
    float acc = bks[c];
    const float* w = Wks + (size_t)c * C_TXT;
    for (int j = 0; j < C_TXT; ++j) acc += s[j] * w[j];
    kbias[(size_t)b * C_TXT + c] = acc;
  } else if (c < C_TXT + C_MEL) {
    int cq = c - C_TXT;
    float acc = bqs[cq];
    const float* w = Wqs + (size_t)cq * C_TXT;
    for (int j = 0; j < C_TXT; ++j) acc += s[j] * w[j];
    qbias[(size_t)b * C_MEL + cq] = acc;
  }
}

// ---------------------------------------------------------------------------
// Tiled transpose + bias + f16 convert:  Y[b][t][c] = f16(X[b][c][t] + bias[b][c])
// ---------------------------------------------------------------------------
__global__ void __launch_bounds__(256)
bias_cvt_t_kernel(const float* __restrict__ X, const float* __restrict__ bias,
                  _Float16* __restrict__ Y, int C, int T) {
  __shared__ float tile[32][33];
  const int b  = blockIdx.z;
  const int t0 = blockIdx.x * 32;
  const int c0 = blockIdx.y * 32;
  const int tx = threadIdx.x & 31;
  const int ty = threadIdx.x >> 5;   // 0..7
  for (int r = ty; r < 32; r += 8) {
    int c = c0 + r, t = t0 + tx;
    tile[r][tx] = (c < C && t < T) ? X[(size_t)b * C * T + (size_t)c * T + t] : 0.0f;
  }
  __syncthreads();
  for (int r = ty; r < 32; r += 8) {
    int t = t0 + r, c = c0 + tx;
    if (t < T && c < C)
      Y[(size_t)b * T * C + (size_t)t * C + c] =
          (_Float16)(tile[tx][r] + bias[(size_t)b * C + c]);
  }
}

// ---------------------------------------------------------------------------
// Weight reorder + f16 convert:  W (O, I, taps) -> Ah[m][tap*I + c]
// ---------------------------------------------------------------------------
__global__ void wcvt_kernel(const float* __restrict__ W, _Float16* __restrict__ Ah,
                            int O, int I, int taps) {
  long i = (long)blockIdx.x * blockDim.x + threadIdx.x;
  long total = (long)O * I * taps;
  if (i >= total) return;
  int m   = (int)(i / ((long)I * taps));
  int rem = (int)(i % ((long)I * taps));
  int tap = rem / I;
  int c   = rem % I;
  Ah[(size_t)m * I * taps + (size_t)tap * I + c] = (_Float16)W[((size_t)m * I + c) * taps + tap];
}

// ---------------------------------------------------------------------------
// WMMA fragment helper: per-lane A/B fragment = two contiguous 8-half runs
// of one LDS row (ISA 7.12.2 16-bit layout): K = kb..kb+7 and kb+16..kb+23.
// ---------------------------------------------------------------------------
__device__ __forceinline__ v16h load_frag_row(const _Float16* row, int kb) {
  v8h lo = *(const v8h*)(row + kb);
  v8h hi = *(const v8h*)(row + kb + 16);
  return __builtin_shufflevector(lo, hi, 0, 1, 2, 3, 4, 5, 6, 7,
                                 8, 9, 10, 11, 12, 13, 14, 15);
}

#define WMMA_F16(A_, B_, C_) \
  __builtin_amdgcn_wmma_f32_16x16x32_f16(false, (A_), false, (B_), (short)0, (C_), false, false)

// ---------------------------------------------------------------------------
// Generic WMMA GEMM with implicit im2col conv, double-buffered async LDS.
//   Out[b][t][m] = act( bias[m] + sum_k A[m][k] * Bact[b][t + k/Kc - taps/2][k%Kc] )
// Block: 256 threads (8 waves), tile 64(M) x 128(N=t), K stepped by 32.
// Wave grid 2(M) x 4(N); each wave computes 32x32 = 4 accumulators.
// tap/channel of the B chunk maintained incrementally (no divisions in loop).
// ---------------------------------------------------------------------------
__global__ void __launch_bounds__(256)
gemm_conv_kernel(const _Float16* __restrict__ A,     // [M][Kc*taps] row-major
                 const _Float16* __restrict__ Bact,  // [B][T][Kc]  t-major
                 const float* __restrict__ bias,     // [M]
                 _Float16* __restrict__ Out,         // [B][T][M]   t-major
                 int M, int Kc, int taps, int T, int relu) {
  __shared__ __attribute__((aligned(16))) _Float16 As[2][64][40];   // [buf][M][K]
  __shared__ __attribute__((aligned(16))) _Float16 Bs[2][128][40];  // [buf][N][K]

  const int b    = blockIdx.z;
  const int n0   = blockIdx.x * 128;
  const int m0   = blockIdx.y * 64;
  const int tid  = threadIdx.x;
  const int lane = tid & 31;
  const int wave = tid >> 5;
  const int wm   = wave & 1;   // M sub-tile (32 rows)
  const int wn   = wave >> 1;  // N sub-tile (32 cols)
  const int Ktot = Kc * taps;
  const int ksteps = (Ktot + 31) >> 5;
  const int hrad = taps >> 1;

  const _Float16* Bb = Bact + (size_t)b * T * Kc;

  v8f acc00 = {}, acc01 = {}, acc10 = {}, acc11 = {};

  const int l16   = lane & 15;
  const int kb8   = (lane >> 4) << 3;   // 0 or 8
  const int arow0 = wm * 32 + l16;
  const int bcol0 = wn * 32 + l16;

  // staging coords: one aligned 8-half chunk per thread for A (64x4 = 256),
  // two for B (128x4 = 512); Ktot % 8 == 0 and Kc % 8 == 0 always.
  const int srow = tid >> 2;        // 0..63
  const int skc  = (tid & 3) * 8;   // 0,8,16,24
  const bool mvalid = (m0 + srow) < M;
  const _Float16* Arow = A + (size_t)(m0 + srow) * Ktot;

  // stage k-step ksx into buffer buf; (tap,c) = decomposition of k = ksx*32+skc
  auto stage = [&](int ksx, int buf, int tap, int c) {
    int k = ksx * 32 + skc;
    _Float16* aDst = &As[buf][srow][skc];
    if (mvalid && k < Ktot) cp_b128(aDst, Arow + k);
    else                    zero_b128(aDst);
    _Float16* bDst0 = &Bs[buf][srow][skc];
    _Float16* bDst1 = &Bs[buf][srow + 64][skc];
    if (tap < taps) {
      int t = n0 + srow + tap - hrad;
      if (t >= 0 && t < T) cp_b128(bDst0, Bb + (size_t)t * Kc + c);
      else                 zero_b128(bDst0);
      t += 64;
      if (t >= 0 && t < T) cp_b128(bDst1, Bb + (size_t)t * Kc + c);
      else                 zero_b128(bDst1);
    } else {
      zero_b128(bDst0);
      zero_b128(bDst1);
    }
  };

  // incremental tap/channel for this thread's B chunk (skc < 32 <= Kc)
  int tapB = 0, cB = skc;

  stage(0, 0, 0, skc);
  wait_async();
  __syncthreads();

  for (int ks = 0; ks < ksteps; ++ks) {
    const int cur = ks & 1, nxt = cur ^ 1;
    const bool have_next = (ks + 1) < ksteps;

    // advance B chunk state (at most one tap boundary per 32-step)
    int ntap = tapB, nc = cB + 32;
    if (nc >= Kc) { nc -= Kc; ++ntap; }

    // issue next tile's async DMA early; overlaps with the WMMAs below
    if (have_next) stage(ks + 1, nxt, ntap, nc);

    // compute from current buffer: 8 b128 LDS loads -> 4 WMMAs
    v16h a0 = load_frag_row(&As[cur][arow0][0], kb8);
    v16h a1 = load_frag_row(&As[cur][arow0 + 16][0], kb8);
    v16h b0 = load_frag_row(&Bs[cur][bcol0][0], kb8);
    v16h b1 = load_frag_row(&Bs[cur][bcol0 + 16][0], kb8);
    acc00 = WMMA_F16(a0, b0, acc00);
    acc01 = WMMA_F16(a0, b1, acc01);
    acc10 = WMMA_F16(a1, b0, acc10);
    acc11 = WMMA_F16(a1, b1, acc11);

    tapB = ntap; cB = nc;
    wait_async();     // next buffer's DMA complete (this wave)
    __syncthreads();  // publish to all waves
  }

  // ---- epilogue: t-major output; accumulator r -> consecutive m -> b128 store
  const int mofs = (lane >> 4) << 3;
  const int mb0 = m0 + wm * 32 + mofs;     // 8 consecutive m
  const int mb1 = mb0 + 16;
  const int nA = n0 + wn * 32 + l16;
  const int nB = nA + 16;
  _Float16* Ob = Out + (size_t)b * T * M;

  float bv0[8], bv1[8];
  if (mb0 < M) {
    const float4* bp = (const float4*)(bias + mb0);
    *(float4*)&bv0[0] = bp[0]; *(float4*)&bv0[4] = bp[1];
  }
  if (mb1 < M) {
    const float4* bp = (const float4*)(bias + mb1);
    *(float4*)&bv1[0] = bp[0]; *(float4*)&bv1[4] = bp[1];
  }

#define STORE_TILE(ACC, MB, BV, N)                                   \
  if ((MB) < M && (N) < T) {                                         \
    v8h o;                                                           \
    _Pragma("unroll")                                                \
    for (int r = 0; r < 8; ++r) {                                    \
      float v = (ACC)[r] + (BV)[r];                                  \
      if (relu) v = v > 0.0f ? v : 0.0f;                             \
      o[r] = (_Float16)v;                                            \
    }                                                                \
    *(v8h*)(Ob + (size_t)(N) * M + (MB)) = o;                        \
  }

  STORE_TILE(acc00, mb0, bv0, nA)
  STORE_TILE(acc01, mb0, bv0, nB)
  STORE_TILE(acc10, mb1, bv1, nA)
  STORE_TILE(acc11, mb1, bv1, nB)
#undef STORE_TILE
}

// ---------------------------------------------------------------------------
// sumsq[b][t] = sum_m X[b][t][m]^2   (t-major input, M % 8 == 0)
// ---------------------------------------------------------------------------
__global__ void sumsq_kernel(const _Float16* __restrict__ X, float* __restrict__ out,
                             int M, int T) {
  int b = blockIdx.y;
  int t = blockIdx.x * blockDim.x + threadIdx.x;
  if (t >= T) return;
  const v8h* x = (const v8h*)(X + ((size_t)b * T + t) * M);
  float s = 0.0f;
  for (int ch = 0; ch < M / 8; ++ch) {
    v8h v = x[ch];
#pragma unroll
    for (int j = 0; j < 8; ++j) { float f = (float)v[j]; s += f * f; }
  }
  out[(size_t)b * T + t] = s;
}

// ---------------------------------------------------------------------------
// Fused attention: per (b, 16 t-rows) block.
//   qk = q k^T via WMMA over K=80 (padded 96), raw = -T*(q2+k2-2qk),
//   row LSE over 400, attn_logprob = raw - lse + log(prior+1e-8),
//   attn = masked, prior-weighted softmax.
// q_t: [B][1600][80], k_t: [B][400][80]  (both t-major, rows 16B-aligned)
// ---------------------------------------------------------------------------
__global__ void __launch_bounds__(256)
attn_kernel(const _Float16* __restrict__ q_t, const _Float16* __restrict__ k_t,
            const float* __restrict__ q2s, const float* __restrict__ k2s,
            const float* __restrict__ prior, const unsigned char* __restrict__ mask,
            float* __restrict__ attn_out, float* __restrict__ logp_out) {
  const int KP = 96;  // padded K
  __shared__ __attribute__((aligned(16))) _Float16 Aq[16][KP];    // [m][k]
  __shared__ __attribute__((aligned(16))) _Float16 Bk[128][KP];   // [n][k]
  __shared__ float raw[16][T2];
  __shared__ float red[2][16][17];
  __shared__ float rowmax[16], rowlse[16], rowden[16];

  const int b   = blockIdx.y;
  const int t0  = blockIdx.x * 16;
  const int tid = threadIdx.x;
  const int lane = tid & 31;
  const int wave = tid >> 5;

  const _Float16* qb = q_t + (size_t)b * T1 * C_ATT;
  const _Float16* kb = k_t + (size_t)b * T2 * C_ATT;

  // stage A = q rows (16 x 96, zero-padded K): 192 chunks of 8 halves, async
  if (tid < 192) {
    int m = tid / 12, kc = (tid % 12) * 8;
    _Float16* dst = &Aq[m][kc];
    if (kc < C_ATT) cp_b128(dst, qb + ((size_t)(t0 + m)) * C_ATT + kc);
    else            zero_b128(dst);
  }

  const int l16 = lane & 15;
  const int kb8 = (lane >> 4) << 3;
  const int mofs = (lane >> 4) << 3;

  // 400 cols = 25 tiles of 16; 4 groups of 128 cols, one 16-col tile per wave
  for (int g = 0; g < 4; ++g) {
    const int nbase = g * 128;
    __syncthreads();  // previous group's Bk fully consumed
    for (int idx = tid; idx < 128 * 12; idx += 256) {  // 6 chunks per thread
      int nn = idx / 12, kc = (idx % 12) * 8;
      int n = nbase + nn;
      _Float16* dst = &Bk[nn][kc];
      if (n < T2 && kc < C_ATT) cp_b128(dst, kb + (size_t)n * C_ATT + kc);
      else                      zero_b128(dst);
    }
    wait_async();     // Aq (g==0) + this group's Bk DMA complete (this wave)
    __syncthreads();  // publish

    const int n0w = nbase + wave * 16;
    if (n0w < T2) {
      v8f acc = {};
#pragma unroll
      for (int ks = 0; ks < 3; ++ks) {
        v16h af = load_frag_row(&Aq[l16][0], ks * 32 + kb8);
        v16h bf = load_frag_row(&Bk[wave * 16 + l16][0], ks * 32 + kb8);
        acc = WMMA_F16(af, bf, acc);
      }
      const int n = n0w + l16;
      const float k2v = k2s[(size_t)b * T2 + n];
#pragma unroll
      for (int r = 0; r < 8; ++r) {
        int m = mofs + r;
        float q2v = q2s[(size_t)b * T1 + t0 + m];
        raw[m][n] = -TEMP * (q2v + k2v - 2.0f * acc[r]);
      }
    }
  }
  __syncthreads();

  // --- row reductions: 16 threads per row, 25 columns each ---
  const int rw = tid >> 4;
  const int p  = tid & 15;
  const int s0 = p * 25;

  float mx = -1e30f;
  for (int s = s0; s < s0 + 25; ++s) mx = fmaxf(mx, raw[rw][s]);
  red[0][rw][p] = mx;
  __syncthreads();
  if (p == 0) {
    float m2 = red[0][rw][0];
    for (int j = 1; j < 16; ++j) m2 = fmaxf(m2, red[0][rw][j]);
    rowmax[rw] = m2;
  }
  __syncthreads();
  mx = rowmax[rw];

  const float* prow = prior + ((size_t)b * T1 + t0 + rw) * T2;
  const unsigned char* mrow = mask + (size_t)b * T2;

  float sAll = 0.0f, sMask = 0.0f;
  for (int s = s0; s < s0 + 25; ++s) {
    float e = __expf(raw[rw][s] - mx);
    sAll += e;
    if (!mrow[s]) sMask += e * (prow[s] + 1e-8f);
  }
  red[0][rw][p] = sAll;
  red[1][rw][p] = sMask;
  __syncthreads();
  if (p == 0) {
    float a = 0.0f, d = 0.0f;
    for (int j = 0; j < 16; ++j) { a += red[0][rw][j]; d += red[1][rw][j]; }
    rowlse[rw] = mx + __logf(a);
    rowden[rw] = d;
  }
  __syncthreads();

  const float lse = rowlse[rw];
  const float inv_den = 1.0f / rowden[rw];
  float* arow = attn_out + ((size_t)b * T1 + t0 + rw) * T2;
  float* lrow = logp_out + ((size_t)b * T1 + t0 + rw) * T2;
  for (int s = s0; s < s0 + 25; ++s) {
    float pv = prow[s] + 1e-8f;
    float rv = raw[rw][s];
    lrow[s] = rv - lse + __logf(pv);
    arow[s] = mrow[s] ? 0.0f : __expf(rv - mx) * pv * inv_den;
  }
}

// ---------------------------------------------------------------------------
// Host launch
// ---------------------------------------------------------------------------
extern "C" void kernel_launch(void* const* d_in, const int* in_sizes, int n_in,
                              void* d_out, int out_size, void* d_ws, size_t ws_size,
                              hipStream_t stream) {
  (void)in_sizes; (void)n_in; (void)out_size; (void)ws_size;

  const float* queries  = (const float*)d_in[0];
  const float* keys     = (const float*)d_in[1];
  const unsigned char* mask = (const unsigned char*)d_in[2]; // jax bool = 1 byte
  const float* prior    = (const float*)d_in[3];
  const float* spk      = (const float*)d_in[4];
  const float* Wk1 = (const float*)d_in[5];   const float* bk1 = (const float*)d_in[6];
  const float* Wk2 = (const float*)d_in[7];   const float* bk2 = (const float*)d_in[8];
  const float* Wq1 = (const float*)d_in[9];   const float* bq1 = (const float*)d_in[10];
  const float* Wq2 = (const float*)d_in[11];  const float* bq2 = (const float*)d_in[12];
  const float* Wq3 = (const float*)d_in[13];  const float* bq3 = (const float*)d_in[14];
  const float* Wks = (const float*)d_in[15];  const float* bks = (const float*)d_in[16];
  const float* Wqs = (const float*)d_in[17];  const float* bqs = (const float*)d_in[18];

  float* attn_out = (float*)d_out;                       // [32][1600][400]
  float* logp_out = attn_out + (size_t)BATCH * T1 * T2;  // second tuple element

  // ---- workspace carve (~87 MB) ----
  char* ws = (char*)d_ws;
  size_t off = 0;
  auto carve = [&](size_t bytes) -> void* {
    void* p = ws + off;
    off = (off + bytes + 255) & ~(size_t)255;
    return p;
  };
  float*     kbias   = (float*)carve((size_t)BATCH * C_TXT * 4);
  float*     qbias   = (float*)carve((size_t)BATCH * C_MEL * 4);
  _Float16*  Wk1h    = (_Float16*)carve((size_t)1024 * 1536 * 2);
  _Float16*  Wk2h    = (_Float16*)carve((size_t)80 * 1024 * 2);
  _Float16*  Wq1h    = (_Float16*)carve((size_t)160 * 240 * 2);
  _Float16*  Wq2h    = (_Float16*)carve((size_t)80 * 160 * 2);
  _Float16*  Wq3h    = (_Float16*)carve((size_t)80 * 80 * 2);
  _Float16*  h_keys  = (_Float16*)carve((size_t)BATCH * T2 * C_TXT * 2);  // [B][T2][512]
  _Float16*  h_k1    = (_Float16*)carve((size_t)BATCH * T2 * 1024 * 2);   // [B][T2][1024]
  _Float16*  k_h     = (_Float16*)carve((size_t)BATCH * T2 * 80 * 2);     // [B][T2][80]
  float*     k2s     = (float*)carve((size_t)BATCH * T2 * 4);
  _Float16*  h_qry   = (_Float16*)carve((size_t)BATCH * T1 * C_MEL * 2);  // [B][T1][80]
  _Float16*  h_q1    = (_Float16*)carve((size_t)BATCH * T1 * 160 * 2);    // [B][T1][160]
  _Float16*  h_q2    = (_Float16*)carve((size_t)BATCH * T1 * 80 * 2);     // [B][T1][80]
  _Float16*  q_h     = (_Float16*)carve((size_t)BATCH * T1 * 80 * 2);     // [B][T1][80]
  float*     q2s     = (float*)carve((size_t)BATCH * T1 * 4);

  // ---- weight reorder + f16 conversion ----
  auto wcvt = [&](const float* W, _Float16* Ah, int O, int I, int taps) {
    long total = (long)O * I * taps;
    wcvt_kernel<<<dim3((unsigned)((total + 255) / 256)), 256, 0, stream>>>(W, Ah, O, I, taps);
  };
  wcvt(Wk1, Wk1h, 1024, 512, 3);
  wcvt(Wk2, Wk2h, 80, 1024, 1);
  wcvt(Wq1, Wq1h, 160, 80, 3);
  wcvt(Wq2, Wq2h, 80, 160, 1);
  wcvt(Wq3, Wq3h, 80, 80, 1);

  // ---- speaker projections ----
  spk_proj_kernel<<<dim3((C_TXT + C_MEL + 127) / 128, BATCH), 128, 0, stream>>>(
      spk, Wks, bks, Wqs, bqs, kbias, qbias);

  // ---- bias add + transpose to t-major + f16 convert ----
  bias_cvt_t_kernel<<<dim3((T2 + 31) / 32, (C_TXT + 31) / 32, BATCH), 256, 0, stream>>>(
      keys, kbias, h_keys, C_TXT, T2);
  bias_cvt_t_kernel<<<dim3((T1 + 31) / 32, (C_MEL + 31) / 32, BATCH), 256, 0, stream>>>(
      queries, qbias, h_qry, C_MEL, T1);

  // ---- conv stack (WMMA GEMMs, t-major activations) ----
  auto gemm = [&](const _Float16* A, const _Float16* Bact, const float* bias,
                  _Float16* Out, int M, int Kc, int taps, int T, int relu) {
    dim3 grid((T + 127) / 128, (M + 63) / 64, BATCH);
    gemm_conv_kernel<<<grid, 256, 0, stream>>>(A, Bact, bias, Out, M, Kc, taps, T, relu);
  };
  // keys path
  gemm(Wk1h, h_keys, bk1, h_k1, 1024, 512, 3, T2, 1);
  gemm(Wk2h, h_k1,   bk2, k_h,  80,  1024, 1, T2, 0);
  // queries path
  gemm(Wq1h, h_qry,  bq1, h_q1, 160, 80,  3, T1, 1);
  gemm(Wq2h, h_q1,   bq2, h_q2, 80,  160, 1, T1, 1);
  gemm(Wq3h, h_q2,   bq3, q_h,  80,  80,  1, T1, 0);

  // ---- squared norms ----
  sumsq_kernel<<<dim3((T2 + 127) / 128, BATCH), 128, 0, stream>>>(k_h, k2s, 80, T2);
  sumsq_kernel<<<dim3((T1 + 127) / 128, BATCH), 128, 0, stream>>>(q_h, q2s, 80, T1);

  // ---- fused attention (qk WMMA + log-softmax + prior + masked softmax) ----
  attn_kernel<<<dim3(T1 / 16, BATCH), 256, 0, stream>>>(
      q_h, k_h, q2s, k2s, prior, mask, attn_out, logp_out);
}
